// SelfAttention_16295105921480
// MI455X (gfx1250) — matrix operand order repaired and checked by reading, not compiled, not run on hardware
//
#include <hip/hip_runtime.h>

typedef __bf16 bf16_t;
typedef __attribute__((ext_vector_type(16))) __bf16 v16bf;
typedef __attribute__((ext_vector_type(8)))  __bf16 v8bf;
typedef __attribute__((ext_vector_type(4)))  __bf16 v4bf;
typedef __attribute__((ext_vector_type(8)))  float  v8f;
typedef __attribute__((ext_vector_type(4)))  unsigned int v4u;
typedef __attribute__((ext_vector_type(8)))  int v8i;
typedef __attribute__((ext_vector_type(4)))  int v4i;

#if __has_builtin(__builtin_amdgcn_tensor_load_to_lds) && \
    __has_builtin(__builtin_amdgcn_s_wait_tensorcnt)
#define USE_TDM 1
#endif

#define WMMA_BF16(a, b, c) \
  __builtin_amdgcn_wmma_f32_16x16x32_bf16(false, (a), false, (b), (short)0, (c), false, false)

// ---------------------------------------------------------------------------
// Fragment loaders (generic pointers: work for global and LDS).
// A operand (16x32 bf16): lane<16 holds row r, K = {0..7, 16..23};
//                         lane>=16 holds row r, K = {8..15, 24..31}.
// B operand (32x16 bf16): lane holds col n=lane%16; lanes<16: K={0..7, 16..23},
//                         lanes>=16: K={8..15, 24..31}.
// ---------------------------------------------------------------------------
__device__ __forceinline__ v16bf ld_fragA(const bf16_t* p, int stride, int row,
                                          int kbase, int koff8) {
  const bf16_t* q = p + (size_t)row * stride + kbase + koff8;
  v16bf f;
  ((v8bf*)&f)[0] = *(const v8bf*)(q);
  ((v8bf*)&f)[1] = *(const v8bf*)(q + 16);
  return f;
}
__device__ __forceinline__ v16bf ld_fragB(const bf16_t* p, int stride, int row,
                                          int kbase, int half) {
  const bf16_t* q = p + (size_t)row * stride + kbase + half * 8;
  v16bf f;
  ((v8bf*)&f)[0] = *(const v8bf*)(q);
  ((v8bf*)&f)[1] = *(const v8bf*)(q + 16);
  return f;
}

// ---------------------------------------------------------------------------
// LayerNorm over D=1024, output bf16. One block (256 thr) per row.
// ---------------------------------------------------------------------------
__global__ __launch_bounds__(256)
void ln_kernel(const float* __restrict__ x, const float* __restrict__ g,
               const float* __restrict__ b, bf16_t* __restrict__ out) {
  const int D = 1024;
  int row = blockIdx.x;
  int t = threadIdx.x;
  const float* xr = x + (size_t)row * D;
  __shared__ float red[256];

  float4 v = ((const float4*)xr)[t];
  red[t] = v.x + v.y + v.z + v.w;
  __syncthreads();
  for (int off = 128; off > 0; off >>= 1) {
    if (t < off) red[t] += red[t + off];
    __syncthreads();
  }
  float mean = red[0] * (1.0f / 1024.0f);
  __syncthreads();

  float d0 = v.x - mean, d1 = v.y - mean, d2 = v.z - mean, d3 = v.w - mean;
  red[t] = d0 * d0 + d1 * d1 + d2 * d2 + d3 * d3;
  __syncthreads();
  for (int off = 128; off > 0; off >>= 1) {
    if (t < off) red[t] += red[t + off];
    __syncthreads();
  }
  float rstd = rsqrtf(red[0] * (1.0f / 1024.0f) + 1e-5f);

  float4 gg = ((const float4*)g)[t];
  float4 bb = ((const float4*)b)[t];
  v4bf o;
  o[0] = (bf16_t)(d0 * rstd * gg.x + bb.x);
  o[1] = (bf16_t)(d1 * rstd * gg.y + bb.y);
  o[2] = (bf16_t)(d2 * rstd * gg.z + bb.z);
  o[3] = (bf16_t)(d3 * rstd * gg.w + bb.w);
  *(v4bf*)(out + (size_t)row * D + t * 4) = o;
}

// ---------------------------------------------------------------------------
// Weight transpose + cast: in f32 [K,N] row-major -> out bf16 [N,K] row-major.
// ---------------------------------------------------------------------------
__global__ __launch_bounds__(256)
void wcast_kernel(const float* __restrict__ in, bf16_t* __restrict__ out,
                  int K, int N) {
  size_t i = (size_t)blockIdx.x * 256 + threadIdx.x;
  size_t total = (size_t)K * N;
  if (i >= total) return;
  int k = (int)(i / N);
  int n = (int)(i % N);
  out[(size_t)n * K + k] = (bf16_t)in[i];
}

// ---------------------------------------------------------------------------
// Tiled bf16 WMMA GEMM: out[M,N] = A[M,K] @ Wt[N,K]^T + bias, fused epilogue.
// Block tile 128x64, BK=64, 256 threads = 8 waves; each wave does 32x32.
// B (weight) tile is staged global->LDS by the Tensor Data Mover (wave 0),
// A tile by vector loads. Dims are multiples of the tile for this problem.
// ---------------------------------------------------------------------------
__global__ __launch_bounds__(256)
void gemm_bf16_kernel(const bf16_t* __restrict__ A, const bf16_t* __restrict__ Wt,
                      const float* __restrict__ bias, const float* __restrict__ resid,
                      float* __restrict__ outF, bf16_t* __restrict__ outB,
                      int Mdim, int Ndim, int Kdim, int relu) {
  const int BK = 64;
  __shared__ __align__(16) bf16_t As[128 * 64];
  __shared__ __align__(16) bf16_t Bs[64 * 64];

  int t = threadIdx.x;
  int m0 = blockIdx.x * 128;
  int n0 = blockIdx.y * 64;
  int wave = t >> 5, lane = t & 31;
  int wm = (wave & 3) * 32;   // wave M offset within block tile
  int wn = (wave >> 2) * 32;  // wave N offset within block tile
  int half = lane >> 4, r = lane & 15;
  int koff8 = half * 8;

#ifdef USE_TDM
  // D# group 1 (constant across the K loop): data_size=2B, tensor_dim0=Kdim,
  // tensor_dim1=64 rows, tile 64x64, dim0 stride = Kdim elements.
  unsigned td0 = (unsigned)Kdim;
  unsigned td1 = 64u;
  unsigned st0 = (unsigned)Kdim;
  v8i dg1 = { (int)0x00010000u,                                    // data_size=1 (2B)
              (int)((td0 & 0xFFFFu) << 16),                        // dim0[15:0]
              (int)(((td0 >> 16) & 0xFFFFu) | ((td1 & 0xFFFFu) << 16)),
              (int)(((td1 >> 16) & 0xFFFFu) | (64u << 16)),        // tile_dim0=64
              (int)64u,                                            // tile_dim1=64
              (int)st0,                                            // dim0_stride lo
              0, 0 };
  v4i dgz = {0, 0, 0, 0};
  unsigned ldsB = (unsigned)(unsigned long long)(void*)Bs;
#endif

  v8f acc[2][2];
#pragma unroll
  for (int fi = 0; fi < 2; fi++)
#pragma unroll
    for (int fj = 0; fj < 2; fj++)
#pragma unroll
      for (int g = 0; g < 8; g++) acc[fi][fj][g] = 0.0f;

  for (int kt = 0; kt < Kdim; kt += BK) {
    if (kt + BK < Kdim) {
      // hint next A tile into cache (emits global_prefetch)
      __builtin_prefetch(A + (size_t)(m0 + (t >> 1)) * Kdim + kt + BK, 0, 1);
    }

    // ---- stage B (weight) tile [64 x 64] ----
#ifdef USE_TDM
    if (wave == 0) {
      unsigned long long ga =
          (unsigned long long)(const void*)(Wt + (size_t)n0 * Kdim + kt);
      v4u dg0 = { 1u,                                     // count=1
                  ldsB,                                   // lds_addr
                  (unsigned)(ga & 0xFFFFFFFFu),           // global_addr[31:0]
                  (unsigned)((ga >> 32) & 0x01FFFFFFu) | 0x80000000u }; // +type=2
#if defined(__clang_major__) && __clang_major__ >= 23
      v8i dz8 = {0, 0, 0, 0, 0, 0, 0, 0};
      __builtin_amdgcn_tensor_load_to_lds(dg0, dg1, dgz, dgz, dz8, 0);
#else
      __builtin_amdgcn_tensor_load_to_lds(dg0, dg1, dgz, dgz, 0);
#endif
      __builtin_amdgcn_s_wait_tensorcnt(0);
    }
#else
#pragma unroll
    for (int i = 0; i < 2; i++) {
      int c = t + i * 256;
      int row = c >> 3, q = c & 7;
      *(v8bf*)&Bs[row * 64 + q * 8] =
          *(const v8bf*)&Wt[(size_t)(n0 + row) * Kdim + kt + q * 8];
    }
#endif

    // ---- stage A tile: 128 rows x 64 bf16 -> 1024 chunks of 8 bf16 ----
#pragma unroll
    for (int i = 0; i < 4; i++) {
      int c = t + i * 256;
      int row = c >> 3, q = c & 7;
      *(v8bf*)&As[row * 64 + q * 8] =
          *(const v8bf*)&A[(size_t)(m0 + row) * Kdim + kt + q * 8];
    }
    __syncthreads();

#pragma unroll
    for (int ks = 0; ks < BK; ks += 32) {
      v16bf a0 = ld_fragA(As, 64, wm + r, ks, koff8);
      v16bf a1 = ld_fragA(As, 64, wm + 16 + r, ks, koff8);
      v16bf b0 = ld_fragB(Bs, 64, wn + r, ks, half);
      v16bf b1 = ld_fragB(Bs, 64, wn + 16 + r, ks, half);
      acc[0][0] = WMMA_BF16(a0, b0, acc[0][0]);
      acc[0][1] = WMMA_BF16(a0, b1, acc[0][1]);
      acc[1][0] = WMMA_BF16(a1, b0, acc[1][0]);
      acc[1][1] = WMMA_BF16(a1, b1, acc[1][1]);
    }
    __syncthreads();
  }

  // Epilogue. C layout: VGPR g -> row g (lanes 0-15) / g+8 (lanes 16-31),
  // col = lane%16.
#pragma unroll
  for (int fi = 0; fi < 2; fi++) {
#pragma unroll
    for (int fj = 0; fj < 2; fj++) {
      int col = n0 + wn + fj * 16 + r;
      float bc = bias ? bias[col] : 0.0f;
#pragma unroll
      for (int g = 0; g < 8; g++) {
        int row = m0 + wm + fi * 16 + half * 8 + g;
        float v = acc[fi][fj][g] + bc;
        if (relu) v = fmaxf(v, 0.0f);
        if (resid) v += resid[(size_t)row * Ndim + col];
        if (outF) outF[(size_t)row * Ndim + col] = v;
        else      outB[(size_t)row * Ndim + col] = (bf16_t)v;
      }
    }
  }
}

// ---------------------------------------------------------------------------
// Flash attention, one wave per (b, h, 16-query tile). Key tile = 32.
// Q,K,V,O are bf16 [B*S, D] with head h occupying cols [h*64, h*64+64).
// ---------------------------------------------------------------------------
__global__ __launch_bounds__(32)
void attn_kernel(const bf16_t* __restrict__ Q, const bf16_t* __restrict__ Kt,
                 const bf16_t* __restrict__ V, const int* __restrict__ mask,
                 bf16_t* __restrict__ O) {
  const int S = 2048, Dm = 1024, HD = 64;
  int qt = blockIdx.x, h = blockIdx.y, b = blockIdx.z;
  int lane = threadIdx.x;
  int half = lane >> 4, r = lane & 15;
  int koff8 = half * 8;

  size_t base = (size_t)b * S * Dm + (size_t)h * HD;
  const bf16_t* qp = Q + base;
  const bf16_t* kp = Kt + base;
  const bf16_t* vp = V + base;

  // Q fragments (rows qt*16 + r, hd 0..31 and 32..63), loaded once.
  int qrow = qt * 16 + r;
  v16bf qa0 = ld_fragA(qp, Dm, qrow, 0, koff8);
  v16bf qa1 = ld_fragA(qp, Dm, qrow, 32, koff8);

  v8f o[4];
  float mrun[8], lrun[8];
#pragma unroll
  for (int g = 0; g < 8; g++) { mrun[g] = -1e30f; lrun[g] = 0.0f; }
#pragma unroll
  for (int f = 0; f < 4; f++)
#pragma unroll
    for (int g = 0; g < 8; g++) o[f][g] = 0.0f;

  __shared__ __align__(16) bf16_t Pl[16 * 32];   // P tile, row-major [16,32]
  __shared__ __align__(16) bf16_t Vt[64 * 32];   // V tile transposed [hd, key]

  const float scale = 0.125f;  // 1/sqrt(64)

  for (int j0 = 0; j0 < S; j0 += 32) {
    // Stage V^T: each lane owns key row j0+lane (64 bf16), scatters to Vt.
    {
      const bf16_t* vr = vp + (size_t)(j0 + lane) * Dm;
#pragma unroll
      for (int d = 0; d < 64; d += 8) {
        v8bf vv = *(const v8bf*)&vr[d];
#pragma unroll
        for (int e = 0; e < 8; e++) Vt[(d + e) * 32 + lane] = vv[e];
      }
    }

    // Scores for two 16-key sub-tiles; B operand = rows of K (cols of K^T).
    v8f s0, s1;
#pragma unroll
    for (int g = 0; g < 8; g++) { s0[g] = 0.0f; s1[g] = 0.0f; }
    {
      v16bf kb;
      kb = ld_fragB(kp, Dm, j0 + r, 0, half);       s0 = WMMA_BF16(qa0, kb, s0);
      kb = ld_fragB(kp, Dm, j0 + r, 32, half);      s0 = WMMA_BF16(qa1, kb, s0);
      kb = ld_fragB(kp, Dm, j0 + 16 + r, 0, half);  s1 = WMMA_BF16(qa0, kb, s1);
      kb = ld_fragB(kp, Dm, j0 + 16 + r, 32, half); s1 = WMMA_BF16(qa1, kb, s1);
    }

    // Scale + mask (score col = key index).
    float pen0 = mask[b * S + j0 + r]      ? 0.0f : -1e9f;
    float pen1 = mask[b * S + j0 + 16 + r] ? 0.0f : -1e9f;

    // Online softmax per row; rows live in 16-lane halves, so xor-shuffles
    // with masks 8,4,2,1 reduce exactly over one row.
#pragma unroll
    for (int g = 0; g < 8; g++) {
      float a0 = s0[g] * scale + pen0;
      float a1 = s1[g] * scale + pen1;
      float tmax = fmaxf(a0, a1);
      for (int off = 8; off > 0; off >>= 1)
        tmax = fmaxf(tmax, __shfl_xor(tmax, off, 32));
      float mnew = fmaxf(mrun[g], tmax);
      float corr = __expf(mrun[g] - mnew);
      float p0 = __expf(a0 - mnew);
      float p1 = __expf(a1 - mnew);
      float ps = p0 + p1;
      for (int off = 8; off > 0; off >>= 1)
        ps += __shfl_xor(ps, off, 32);
      lrun[g] = lrun[g] * corr + ps;
      mrun[g] = mnew;
#pragma unroll
      for (int f = 0; f < 4; f++) o[f][g] *= corr;
      int prow = half * 8 + g;
      Pl[prow * 32 + r]      = (bf16_t)p0;
      Pl[prow * 32 + r + 16] = (bf16_t)p1;
    }
    __syncthreads();

    // o += P(16x32) @ Vtile(32x64), four 16-col output fragments.
    v16bf pa = ld_fragA(Pl, 32, r, 0, koff8);
#pragma unroll
    for (int f = 0; f < 4; f++) {
      v16bf vb = ld_fragB(Vt, 32, f * 16 + r, 0, half);
      o[f] = WMMA_BF16(pa, vb, o[f]);
    }
    __syncthreads();
  }

  // Normalize and write attended (bf16).
#pragma unroll
  for (int f = 0; f < 4; f++) {
#pragma unroll
    for (int g = 0; g < 8; g++) {
      int row = qt * 16 + half * 8 + g;
      int col = f * 16 + r;
      float val = o[f][g] / lrun[g];
      O[base + (size_t)row * Dm + col] = (bf16_t)val;
    }
  }
}

// ---------------------------------------------------------------------------
// Host launch
// ---------------------------------------------------------------------------
extern "C" void kernel_launch(void* const* d_in, const int* in_sizes, int n_in,
                              void* d_out, int out_size, void* d_ws, size_t ws_size,
                              hipStream_t stream) {
  const int Bb = 2, S = 2048, D = 1024, F = 4096;
  const int M = Bb * S;  // 4096 token rows

  const float* x    = (const float*)d_in[0];
  const int*   mask = (const int*)d_in[1];
  const float* wq   = (const float*)d_in[2];
  const float* bq   = (const float*)d_in[3];
  const float* wk   = (const float*)d_in[4];
  const float* bk   = (const float*)d_in[5];
  const float* wv   = (const float*)d_in[6];
  const float* bv   = (const float*)d_in[7];
  const float* wo   = (const float*)d_in[8];
  const float* bo   = (const float*)d_in[9];
  const float* g1   = (const float*)d_in[10];
  const float* be1  = (const float*)d_in[11];
  const float* g2   = (const float*)d_in[12];
  const float* be2  = (const float*)d_in[13];
  const float* w1   = (const float*)d_in[14];
  const float* bf1  = (const float*)d_in[15];
  const float* w2   = (const float*)d_in[16];
  const float* bf2  = (const float*)d_in[17];
  float* out = (float*)d_out;

  char* wsp = (char*)d_ws;
  auto alloc = [&](size_t bytes) -> void* {
    void* p = (void*)wsp;
    wsp += (bytes + 255) & ~(size_t)255;
    return p;
  };
  bf16_t* xn1  = (bf16_t*)alloc((size_t)M * D * 2);
  bf16_t* wqT  = (bf16_t*)alloc((size_t)D * D * 2);
  bf16_t* wkT  = (bf16_t*)alloc((size_t)D * D * 2);
  bf16_t* wvT  = (bf16_t*)alloc((size_t)D * D * 2);
  bf16_t* woT  = (bf16_t*)alloc((size_t)D * D * 2);
  bf16_t* w1T  = (bf16_t*)alloc((size_t)F * D * 2);
  bf16_t* w2T  = (bf16_t*)alloc((size_t)D * F * 2);
  bf16_t* qb   = (bf16_t*)alloc((size_t)M * D * 2);
  bf16_t* kb   = (bf16_t*)alloc((size_t)M * D * 2);
  bf16_t* vb   = (bf16_t*)alloc((size_t)M * D * 2);
  bf16_t* att  = (bf16_t*)alloc((size_t)M * D * 2);
  float*  x1   = (float*) alloc((size_t)M * D * 4);
  bf16_t* xn2  = (bf16_t*)alloc((size_t)M * D * 2);
  bf16_t* hmid = (bf16_t*)alloc((size_t)M * F * 2);

  // --- LN1 ---
  ln_kernel<<<M, 256, 0, stream>>>(x, g1, be1, xn1);

  // --- weight transpose + cast ---
  {
    int nDD = (D * D + 255) / 256;
    wcast_kernel<<<nDD, 256, 0, stream>>>(wq, wqT, D, D);
    wcast_kernel<<<nDD, 256, 0, stream>>>(wk, wkT, D, D);
    wcast_kernel<<<nDD, 256, 0, stream>>>(wv, wvT, D, D);
    wcast_kernel<<<nDD, 256, 0, stream>>>(wo, woT, D, D);
    int nDF = (D * F + 255) / 256;
    wcast_kernel<<<nDF, 256, 0, stream>>>(w1, w1T, D, F);   // [D,F] -> [F,D]
    wcast_kernel<<<nDF, 256, 0, stream>>>(w2, w2T, F, D);   // [F,D] -> [D,F]
  }

  // --- QKV projections (bf16 out) ---
  dim3 gDD(M / 128, D / 64);
  gemm_bf16_kernel<<<gDD, 256, 0, stream>>>(xn1, wqT, bq, nullptr, nullptr, qb, M, D, D, 0);
  gemm_bf16_kernel<<<gDD, 256, 0, stream>>>(xn1, wkT, bk, nullptr, nullptr, kb, M, D, D, 0);
  gemm_bf16_kernel<<<gDD, 256, 0, stream>>>(xn1, wvT, bv, nullptr, nullptr, vb, M, D, D, 0);

  // --- flash attention ---
  attn_kernel<<<dim3(S / 16, 16, Bb), 32, 0, stream>>>(qb, kb, vb, mask, att);

  // --- output projection + residual (f32 out) ---
  gemm_bf16_kernel<<<gDD, 256, 0, stream>>>(att, woT, bo, x, x1, nullptr, M, D, D, 0);

  // --- LN2 ---
  ln_kernel<<<M, 256, 0, stream>>>(x1, g2, be2, xn2);

  // --- FFN ---
  dim3 gDF(M / 128, F / 64);
  gemm_bf16_kernel<<<gDF, 256, 0, stream>>>(xn2, w1T, bf1, nullptr, nullptr, hmid, M, F, D, 1);
  gemm_bf16_kernel<<<gDD, 256, 0, stream>>>(hmid, w2T, bf2, x1, out, nullptr, M, D, F, 0);
}
